// CausalMultiHeadAttention_91319594648274
// MI455X (gfx1250) — matrix operand order repaired and checked by reading, not compiled, not run on hardware
//
#include <hip/hip_runtime.h>
#include <hip/hip_bf16.h>

// ---------------------------------------------------------------------------
// Causal MHA for MI455X (gfx1250, wave32, WMMA).
// B=4, S=2048, D=1024, H=16, HD=64.  bf16 WMMA with fp32 accumulation.
// Pipeline: convert -> fused QKV gemm -> flash attention -> out-proj gemm.
// GEMMs: per-wave 32x64 tiles, B tile shared via LDS, async double-buffered.
// Attention: K/V tiles staged to LDS with async copies, online softmax.
// ---------------------------------------------------------------------------

#define BB 4
#define SS 2048
#define DD 1024
#define HH 16
#define HD 64

typedef __attribute__((ext_vector_type(16))) __bf16 v16bf;
typedef __attribute__((ext_vector_type(8)))  __bf16 v8bf;
typedef __attribute__((ext_vector_type(8)))  float  v8f;

#ifndef __has_builtin
#define __has_builtin(x) 0
#endif

#if __has_builtin(__builtin_amdgcn_global_load_async_to_lds_b128)
#define HAVE_ASYNC_LDS 1
#else
#define HAVE_ASYNC_LDS 0
#endif

// The async builtin is typed as (global int4*, local int4*, imm off, imm cpol)
typedef int v4i32 __attribute__((vector_size(16)));
typedef __attribute__((address_space(1))) v4i32* gb128p;
typedef __attribute__((address_space(3))) v4i32* lb128p;

// 16-byte global -> LDS copy; async (ASYNCcnt-tracked) when available.
static __device__ __forceinline__ void async_cp16(void* ldst, const void* gsrc) {
#if HAVE_ASYNC_LDS
  __builtin_amdgcn_global_load_async_to_lds_b128((gb128p)gsrc, (lb128p)ldst, 0, 0);
#else
  *(v8bf*)ldst = *(const v8bf*)gsrc;
#endif
}

static __device__ __forceinline__ void async_wait0() {
#if HAVE_ASYNC_LDS
#if __has_builtin(__builtin_amdgcn_s_wait_asynccnt)
  __builtin_amdgcn_s_wait_asynccnt(0);
#else
  asm volatile("s_wait_asynccnt 0x0" ::: "memory");
#endif
#endif
}

// wait until only the most-recently-issued tile (N chunks/thread) is pending
static __device__ __forceinline__ void async_wait_le2() {
#if HAVE_ASYNC_LDS
#if __has_builtin(__builtin_amdgcn_s_wait_asynccnt)
  __builtin_amdgcn_s_wait_asynccnt(2);
#else
  asm volatile("s_wait_asynccnt 0x2" ::: "memory");
#endif
#endif
}

static __device__ __forceinline__ void async_wait_le4() {
#if HAVE_ASYNC_LDS
#if __has_builtin(__builtin_amdgcn_s_wait_asynccnt)
  __builtin_amdgcn_s_wait_asynccnt(4);
#else
  asm volatile("s_wait_asynccnt 0x4" ::: "memory");
#endif
#endif
}

static __device__ __forceinline__ v16bf combine16(v8bf lo, v8bf hi) {
  v16bf r;
#pragma unroll
  for (int i = 0; i < 8; ++i) { r[i] = lo[i]; r[i + 8] = hi[i]; }
  return r;
}

// A-fragment chunk loader: lane l (0-15) = row,
// lane-half selects K sub-block. Caller passes pointer at (row, k0 + half*16);
// elements 0..15 = K k0+half*16 .. k0+half*16+15 (contiguous).
static __device__ __forceinline__ v16bf load_a16(const __hip_bfloat16* p) {
  v8bf lo = *(const v8bf*)p;
  v8bf hi = *(const v8bf*)(p + 8);
  return combine16(lo, hi);
}

// B-fragment loader: lane n (&15) = output column, kbase = (lane>>4)*16,
// elements 0..15 = 16 consecutive K values (column-major/transposed storage).
static __device__ __forceinline__ v16bf load_b16(const __hip_bfloat16* p) {
  v8bf lo = *(const v8bf*)p;
  v8bf hi = *(const v8bf*)(p + 8);
  return combine16(lo, hi);
}

static __device__ __forceinline__ v8f wmma_bf16(v16bf a, v16bf b, v8f c) {
  return __builtin_amdgcn_wmma_f32_16x16x32_bf16(false, a, false, b,
                                                 (short)0, c, false, false);
}

// ------------------------------ converters ---------------------------------

__global__ void cvt_x_kernel(const float* __restrict__ x,
                             __hip_bfloat16* __restrict__ xb, int n) {
  for (int i = blockIdx.x * blockDim.x + threadIdx.x; i < n;
       i += gridDim.x * blockDim.x)
    xb[i] = __float2bfloat16(x[i]);
}

// Store W transposed: wt[n*D + k] = W[k*D + n]  (bf16)
__global__ void cvt_wt_kernel(const float* __restrict__ w,
                              __hip_bfloat16* __restrict__ wt) {
  for (int i = blockIdx.x * blockDim.x + threadIdx.x; i < DD * DD;
       i += gridDim.x * blockDim.x) {
    int n = i >> 10;
    int k = i & (DD - 1);
    wt[i] = __float2bfloat16(w[(size_t)k * DD + n]);
  }
}

// ------------------------- fused QKV projection ----------------------------
// grid = (D/64, (B*S)/128, 3), block = 128 (4 waves). Wave w computes a
// 32x64 tile (rows m0..m0+31). The 64x32 B tile is staged in LDS (shared by
// all 4 waves) with async double buffering.
__global__ __launch_bounds__(128) void qkv_gemm_kernel(
    const __hip_bfloat16* __restrict__ xb,
    const __hip_bfloat16* __restrict__ wt3,   // Wq^T,Wk^T,Wv^T contiguous
    __hip_bfloat16* __restrict__ qout,        // [B,H,S,HD]
    __hip_bfloat16* __restrict__ kout,        // [B,H,S,HD]
    __hip_bfloat16* __restrict__ vtout) {     // [B,H,HD,S]
  __shared__ __attribute__((aligned(16))) __hip_bfloat16 Bs[2][64][32];

  const int tid  = threadIdx.x;
  const int lane = tid & 31;
  const int wave = tid >> 5;
  const int half = lane >> 4;
  const int l16  = lane & 15;
  const int z    = blockIdx.z;
  const int n0   = blockIdx.x * 64;
  const int m0   = blockIdx.y * 128 + wave * 32;

  const __hip_bfloat16* w = wt3 + (size_t)z * (DD * DD);
  const __hip_bfloat16* arow0 = xb + (size_t)(m0 + l16) * DD + half * 16;
  const __hip_bfloat16* arow1 = arow0 + (size_t)16 * DD;

  // stage 64 rows x 32 K of W^T into Bs[buf]: 256 16B chunks, 2 per thread
  auto stage_b = [&](int buf, int k0) {
#pragma unroll
    for (int j = 0; j < 2; ++j) {
      const int c  = tid * 2 + j;          // 0..255
      const int nr = c >> 2;               // 64 rows, 4 chunks each
      const int ko = (c & 3) * 8;
      async_cp16(&Bs[buf][nr][ko], w + (size_t)(n0 + nr) * DD + k0 + ko);
    }
  };

  v8f acc[2][4] = {};
  stage_b(0, 0);
  for (int k0 = 0; k0 < DD; k0 += 32) {
    const int cur = (k0 >> 5) & 1;
    if (k0 + 32 < DD) { stage_b(cur ^ 1, k0 + 32); async_wait_le2(); }
    else              { async_wait0(); }
    __syncthreads();

    const v16bf a0 = load_a16(arow0 + k0);
    const v16bf a1 = load_a16(arow1 + k0);
#pragma unroll
    for (int nt = 0; nt < 4; ++nt) {
      const v16bf bf = load_b16(&Bs[cur][nt * 16 + l16][half * 16]);
      acc[0][nt] = wmma_bf16(a0, bf, acc[0][nt]);
      acc[1][nt] = wmma_bf16(a1, bf, acc[1][nt]);
    }
    __syncthreads();   // reads done before next overwrite of this buffer
  }

#pragma unroll
  for (int a = 0; a < 2; ++a) {
#pragma unroll
    for (int nt = 0; nt < 4; ++nt) {
#pragma unroll
      for (int r = 0; r < 8; ++r) {
        const int m  = m0 + a * 16 + half * 8 + r;   // global row = b*S + s
        const int n  = n0 + nt * 16 + l16;           // global col = h*HD + hd
        const int b  = m >> 11;
        const int s  = m & (SS - 1);
        const int h  = n >> 6;
        const int hd = n & (HD - 1);
        const __hip_bfloat16 v = __float2bfloat16(acc[a][nt][r]);
        if (z == 0)
          qout[(((size_t)b * HH + h) * SS + s) * HD + hd] = v;
        else if (z == 1)
          kout[(((size_t)b * HH + h) * SS + s) * HD + hd] = v;
        else
          vtout[(((size_t)b * HH + h) * HD + hd) * SS + s] = v;
      }
    }
  }
}

// ----------------------------- flash attention -----------------------------
// grid = (B*H, S/64), block = 128 (4 waves). K/V key-tiles (32 keys) are
// async-staged into LDS double buffers shared by all waves; each wave owns 16
// query rows with online softmax. Loop count is uniform per block so
// __syncthreads is legal; surplus tiles are fully causal-masked.
__global__ __launch_bounds__(128) void flash_attn_kernel(
    const __hip_bfloat16* __restrict__ Q,     // [B,H,S,HD]
    const __hip_bfloat16* __restrict__ K,     // [B,H,S,HD]
    const __hip_bfloat16* __restrict__ Vt,    // [B,H,HD,S]
    const unsigned char* __restrict__ mask,   // [B,S] validity
    __hip_bfloat16* __restrict__ ctx) {       // [B,S,D]
  __shared__ __attribute__((aligned(16))) __hip_bfloat16 Ks[2][32][64];
  __shared__ __attribute__((aligned(16))) __hip_bfloat16 Vs[2][64][32];
  __shared__ __attribute__((aligned(16))) __hip_bfloat16 psm[4][16][32];

  const int tid   = threadIdx.x;
  const int lane  = tid & 31;
  const int wave  = tid >> 5;
  const int half  = lane >> 4;
  const int l16   = lane & 15;
  const int bh    = blockIdx.x;
  const int b     = bh >> 4;                  // H == 16
  const int h     = bh & 15;
  const int qblk  = blockIdx.y * 64;
  const int qbase = qblk + wave * 16;

  const __hip_bfloat16* Qh = Q + (size_t)bh * SS * HD;
  const __hip_bfloat16* Kh = K + (size_t)bh * SS * HD;
  const __hip_bfloat16* Vh = Vt + (size_t)bh * HD * SS;
  const unsigned char*  mb = mask + (size_t)b * SS;

  // stage one 32-key tile of K (32x64) and V^T (64x32): 4 chunks per thread
  auto stage_kv = [&](int buf, int key0) {
#pragma unroll
    for (int j = 0; j < 2; ++j) {
      const int c  = tid * 2 + j;            // 0..255
      const int kr = c >> 3;                 // 32 rows x 8 chunks
      const int ko = (c & 7) * 8;
      async_cp16(&Ks[buf][kr][ko], Kh + (size_t)(key0 + kr) * HD + ko);
    }
#pragma unroll
    for (int j = 0; j < 2; ++j) {
      const int c  = tid * 2 + j;
      const int vr = c >> 2;                 // 64 rows x 4 chunks
      const int vo = (c & 3) * 8;
      async_cp16(&Vs[buf][vr][vo], Vh + (size_t)vr * SS + key0 + vo);
    }
  };

  // Q A-fragments held in registers for the whole key loop (HD=64 -> 2 frags)
  const __hip_bfloat16* qrow = Qh + (size_t)(qbase + l16) * HD + half * 16;
  const v16bf qf0 = load_a16(qrow);
  const v16bf qf1 = load_a16(qrow + 32);

  float m_i[8], l_i[8];
#pragma unroll
  for (int r = 0; r < 8; ++r) { m_i[r] = -3.0e38f; l_i[r] = 0.0f; }
  v8f o[4] = {};

  const int ntiles = qblk / 32 + 2;           // keys 0 .. qblk+63 (uniform)
  stage_kv(0, 0);
  for (int jt = 0; jt < ntiles; ++jt) {
    const int key0 = jt * 32;
    const int cur  = jt & 1;
    if (jt + 1 < ntiles) { stage_kv(cur ^ 1, key0 + 32); async_wait_le4(); }
    else                 { async_wait0(); }
    __syncthreads();

    // ---- scores: S = Q * K^T over HD=64, two 16-key column sub-tiles ----
    v8f sc[2];
#pragma unroll
    for (int t = 0; t < 2; ++t) {
      const __hip_bfloat16* krow = &Ks[cur][t * 16 + l16][half * 16];
      v8f c = {};
      c = wmma_bf16(qf0, load_b16(krow), c);
      c = wmma_bf16(qf1, load_b16(krow + 32), c);
      sc[t] = c;
    }

    const int  keyA = key0 + l16;
    const int  keyB = keyA + 16;
    const bool vA   = mb[keyA] != 0;
    const bool vB   = mb[keyB] != 0;

    // ---- online softmax per row (row = half*8 + r, keys across 16 lanes) --
#pragma unroll
    for (int r = 0; r < 8; ++r) {
      const int qrowg = qbase + half * 8 + r;
      float s0 = sc[0][r] * 0.125f;             // 1/sqrt(HD)
      float s1 = sc[1][r] * 0.125f;
      s0 = (vA && keyA <= qrowg) ? s0 : -3.0e38f;
      s1 = (vB && keyB <= qrowg) ? s1 : -3.0e38f;

      float tm = fmaxf(s0, s1);
#pragma unroll
      for (int off = 1; off < 16; off <<= 1)
        tm = fmaxf(tm, __shfl_xor(tm, off, 32));
      const float mnew = fmaxf(m_i[r], tm);

      const float p0 = __expf(s0 - mnew);
      const float p1 = __expf(s1 - mnew);
      float ps = p0 + p1;
#pragma unroll
      for (int off = 1; off < 16; off <<= 1) ps += __shfl_xor(ps, off, 32);

      const float alpha = __expf(m_i[r] - mnew);
      l_i[r] = l_i[r] * alpha + ps;
      m_i[r] = mnew;
#pragma unroll
      for (int i = 0; i < 4; ++i) o[i][r] *= alpha;

      // stage P (C-layout) into per-wave LDS for the layout transpose
      psm[wave][half * 8 + r][l16]      = __float2bfloat16(p0);
      psm[wave][half * 8 + r][16 + l16] = __float2bfloat16(p1);
    }

    // wave is lockstep; drain the DS pipe before cross-lane readback
    asm volatile("s_wait_dscnt 0x0" ::: "memory");

    // ---- P (16x32) re-read as A-fragment; O += P * V ----------------------
    const v16bf pf = load_a16(&psm[wave][l16][half * 16]);
#pragma unroll
    for (int i = 0; i < 4; ++i) {
      o[i] = wmma_bf16(pf, load_b16(&Vs[cur][i * 16 + l16][half * 16]), o[i]);
    }
    __syncthreads();   // all reads of buf `cur` done before it is restaged
  }

  // ---- epilogue: normalize and merge heads into ctx [B,S,D] ---------------
#pragma unroll
  for (int r = 0; r < 8; ++r) {
    const float inv  = l_i[r] > 0.0f ? 1.0f / l_i[r] : 0.0f;
    const int   srow = qbase + half * 8 + r;
#pragma unroll
    for (int i = 0; i < 4; ++i) {
      ctx[((size_t)b * SS + srow) * DD + h * HD + i * 16 + l16] =
          __float2bfloat16(o[i][r] * inv);
    }
  }
}

// --------------------------- output projection -----------------------------
// out = ctx @ Wo + bo, masked. Same LDS-staged 32x64-per-wave structure.
__global__ __launch_bounds__(128) void out_gemm_kernel(
    const __hip_bfloat16* __restrict__ ctx,   // [B*S, D] bf16
    const __hip_bfloat16* __restrict__ wot,   // Wo^T [D, D] bf16
    const float* __restrict__ bo,
    const unsigned char* __restrict__ mask,   // [B*S]
    float* __restrict__ out) {                // [B*S, D] f32
  __shared__ __attribute__((aligned(16))) __hip_bfloat16 Bs[2][64][32];

  const int tid  = threadIdx.x;
  const int lane = tid & 31;
  const int wave = tid >> 5;
  const int half = lane >> 4;
  const int l16  = lane & 15;
  const int n0   = blockIdx.x * 64;
  const int m0   = blockIdx.y * 128 + wave * 32;

  const __hip_bfloat16* arow0 = ctx + (size_t)(m0 + l16) * DD + half * 16;
  const __hip_bfloat16* arow1 = arow0 + (size_t)16 * DD;

  auto stage_b = [&](int buf, int k0) {
#pragma unroll
    for (int j = 0; j < 2; ++j) {
      const int c  = tid * 2 + j;
      const int nr = c >> 2;
      const int ko = (c & 3) * 8;
      async_cp16(&Bs[buf][nr][ko], wot + (size_t)(n0 + nr) * DD + k0 + ko);
    }
  };

  v8f acc[2][4] = {};
  stage_b(0, 0);
  for (int k0 = 0; k0 < DD; k0 += 32) {
    const int cur = (k0 >> 5) & 1;
    if (k0 + 32 < DD) { stage_b(cur ^ 1, k0 + 32); async_wait_le2(); }
    else              { async_wait0(); }
    __syncthreads();

    const v16bf a0 = load_a16(arow0 + k0);
    const v16bf a1 = load_a16(arow1 + k0);
#pragma unroll
    for (int nt = 0; nt < 4; ++nt) {
      const v16bf bf = load_b16(&Bs[cur][nt * 16 + l16][half * 16]);
      acc[0][nt] = wmma_bf16(a0, bf, acc[0][nt]);
      acc[1][nt] = wmma_bf16(a1, bf, acc[1][nt]);
    }
    __syncthreads();
  }

#pragma unroll
  for (int a = 0; a < 2; ++a) {
#pragma unroll
    for (int nt = 0; nt < 4; ++nt) {
#pragma unroll
      for (int r = 0; r < 8; ++r) {
        const int m = m0 + a * 16 + half * 8 + r;
        const int n = n0 + nt * 16 + l16;
        const float v = acc[a][nt][r] + bo[n];
        out[(size_t)m * DD + n] = mask[m] ? v : 0.0f;
      }
    }
  }
}

// ------------------------------- launcher ----------------------------------

extern "C" void kernel_launch(void* const* d_in, const int* in_sizes, int n_in,
                              void* d_out, int out_size, void* d_ws,
                              size_t ws_size, hipStream_t stream) {
  const float* x  = (const float*)d_in[0];
  const float* Wq = (const float*)d_in[1];
  const float* Wk = (const float*)d_in[2];
  const float* Wv = (const float*)d_in[3];
  const float* Wo = (const float*)d_in[4];
  const float* bo = (const float*)d_in[5];
  const unsigned char* mask = (const unsigned char*)d_in[6];

  const size_t NX = (size_t)BB * SS * DD;   // 8 Mi elements
  const size_t NW = (size_t)DD * DD;        // 1 Mi elements

  char* p = (char*)d_ws;
  __hip_bfloat16* xb  = (__hip_bfloat16*)p; p += NX * 2;       // x bf16
  __hip_bfloat16* wt  = (__hip_bfloat16*)p; p += 4 * NW * 2;   // Wq/Wk/Wv/Wo ^T
  __hip_bfloat16* q   = (__hip_bfloat16*)p; p += NX * 2;       // [B,H,S,HD]
  __hip_bfloat16* k   = (__hip_bfloat16*)p; p += NX * 2;       // [B,H,S,HD]
  __hip_bfloat16* vt  = (__hip_bfloat16*)p; p += NX * 2;       // [B,H,HD,S]
  __hip_bfloat16* ctx = (__hip_bfloat16*)p; p += NX * 2;       // [B,S,D]

  cvt_x_kernel<<<2048, 256, 0, stream>>>(x, xb, (int)NX);
  cvt_wt_kernel<<<1024, 256, 0, stream>>>(Wq, wt + 0 * NW);
  cvt_wt_kernel<<<1024, 256, 0, stream>>>(Wk, wt + 1 * NW);
  cvt_wt_kernel<<<1024, 256, 0, stream>>>(Wv, wt + 2 * NW);
  cvt_wt_kernel<<<1024, 256, 0, stream>>>(Wo, wt + 3 * NW);

  qkv_gemm_kernel<<<dim3(DD / 64, (BB * SS) / 128, 3), 128, 0, stream>>>(
      xb, wt, q, k, vt);

  flash_attn_kernel<<<dim3(BB * HH, SS / 64), 128, 0, stream>>>(q, k, vt, mask,
                                                                ctx);

  out_gemm_kernel<<<dim3(DD / 64, (BB * SS) / 128), 128, 0, stream>>>(
      ctx, wt + 3 * NW, bo, mask, (float*)d_out);
}
